// FCGF_point_att_k_fc_89575837925681
// MI455X (gfx1250) — matrix operand order repaired and checked by reading, not compiled, not run on hardware
//
#include <hip/hip_runtime.h>
#include <hip/hip_bf16.h>

#define N_PTS 32768
#define C_IN  32
#define FC0   256
#define KCH   1024
#define BSEG  16
#define BN_EPS 1e-5f

typedef __attribute__((ext_vector_type(16))) __bf16 v16bf;
typedef __attribute__((ext_vector_type(8)))  float  v8f;
typedef __attribute__((ext_vector_type(2)))  float  v2f;
typedef __attribute__((ext_vector_type(8)))  unsigned short ushort8;

union FragBF { v16bf v; ushort8 h[2]; unsigned short u[16]; };

__device__ __forceinline__ unsigned short f2bf_bits(float f) {
  unsigned u = __float_as_uint(f);
  unsigned r = u + 0x7FFFu + ((u >> 16) & 1u);   // round-to-nearest-even
  return (unsigned short)(r >> 16);
}

// A fragment (16-bit, 16x32): lane holds row (lane&15); elems 0-7 -> k0+8*half+j,
// elems 8-15 -> k0+16+8*half+(j-8).  Rows contiguous in memory, stride in elements.
__device__ __forceinline__ v16bf load_afrag(const unsigned short* __restrict__ base,
                                            int stride, int row0, int k0, int lane) {
  const int half = lane >> 4, m = lane & 15;
  const unsigned short* p = base + (size_t)(row0 + m) * stride + k0 + 8 * half;
  FragBF f;
  f.h[0] = *(const ushort8*)p;
  f.h[1] = *(const ushort8*)(p + 16);
  return f.v;
}

// B fragment (16-bit, 32x16) from a weight matrix stored [out,in] (row-major):
// B[k,n] = W[n0+n, k0+k]; lane holds column n=(lane&15); elems 0-7 -> k = 8*half + j,
// elems 8-15 -> k = 16 + 8*half + (j-8).
__device__ __forceinline__ v16bf load_bfrag_w(const unsigned short* __restrict__ w,
                                              int stride, int k0, int n0, int lane) {
  const int half = lane >> 4, n = lane & 15;
  const unsigned short* p = w + (size_t)(n0 + n) * stride + k0 + 8 * half;
  FragBF f;
  f.h[0] = *(const ushort8*)(p);
  f.h[1] = *(const ushort8*)(p + 16);
  return f.v;
}

// ---------------- prep: fold BN into scale/bias, prefix-sum lengths ----------------
__global__ void k_prep(const int* __restrict__ length,
                       const float* b1, const float* g1, const float* be1, const float* m1, const float* v1,
                       const float* b2, const float* g2, const float* be2, const float* m2, const float* v2,
                       const float* fcb, const float* g3, const float* be3, const float* m3, const float* v3,
                       float* scale1, float* bias1, float* scale2, float* bias2,
                       float* scale3, float* bias3, int* off, float* inv_len) {
  const int t = threadIdx.x;  // 1024 threads
  if (t < FC0) {
    float s = g1[t] * rsqrtf(v1[t] + BN_EPS);
    scale1[t] = s;
    bias1[t]  = (b1[t] - m1[t]) * s + be1[t];
    float s3 = g3[t] * rsqrtf(v3[t] + BN_EPS);
    scale3[t] = s3;
    bias3[t]  = (fcb[t] - m3[t]) * s3 + be3[t];   // fcb folded into BN3 bias
  }
  if (t < KCH) {
    float s = g2[t] * rsqrtf(v2[t] + BN_EPS);
    scale2[t] = s;
    bias2[t]  = (b2[t] - m2[t]) * s + be2[t];
  }
  if (t < BSEG) inv_len[t] = 1.0f / (float)length[t];
  if (t == 0) {
    int acc = 0; off[0] = 0;
    for (int i = 0; i < BSEG; ++i) { acc += length[i]; off[i + 1] = acc; }
  }
}

__global__ void k_cvt(const float* __restrict__ src, unsigned short* __restrict__ dst, int n) {
  int i = blockIdx.x * blockDim.x + threadIdx.x;
  if (i < n) dst[i] = f2bf_bits(src[i]);
}

__global__ void k_zero(float* __restrict__ p, int n) {
  int i = blockIdx.x * blockDim.x + threadIdx.x;
  if (i < n) p[i] = 0.0f;
}

// ---------------- GEMM1: h = relu(affine1(x @ w1^T))  (K-reduction = 32 = one WMMA) ----
__global__ void k_gemm1(const unsigned short* __restrict__ xb,
                        const unsigned short* __restrict__ w1b,
                        const float* __restrict__ scale1, const float* __restrict__ bias1,
                        unsigned short* __restrict__ hb) {
  const int lane = threadIdx.x & 31;
  const int wave = threadIdx.x >> 5;
  const int tile = blockIdx.x * 8 + wave;   // (N/16)*(FC0/16) = 32768 tiles
  const int n0 = (tile >> 4) * 16;
  const int f0 = (tile & 15) * 16;
  v16bf a = load_afrag(xb, C_IN, n0, 0, lane);
  v16bf b = load_bfrag_w(w1b, C_IN, 0, f0, lane);
  v8f c = {};
  c = __builtin_amdgcn_wmma_f32_16x16x32_bf16(false, a, false, b, (short)0, c, false, false);
  const int half = lane >> 4, ncol = lane & 15;
  const int f = f0 + ncol;
  const float s = scale1[f], bi = bias1[f];
#pragma unroll
  for (int v = 0; v < 8; ++v) {
    int row = n0 + v + 8 * half;
    float val = fmaxf(c[v] * s + bi, 0.0f);
    hb[(size_t)row * FC0 + f] = f2bf_bits(val);
  }
}

// ---------------- GEMM2: att = relu(affine2(h @ w2^T)) ; 32x64 block per wave --------
// 2 A-frags x 4 B-frags -> 8 WMMAs per k-slice: halves B traffic vs 16x64.
__global__ void k_gemm2(const unsigned short* __restrict__ hb,
                        const unsigned short* __restrict__ w2b,
                        const float* __restrict__ scale2, const float* __restrict__ bias2,
                        float* __restrict__ att) {
  const int lane = threadIdx.x & 31;
  const int wave = threadIdx.x >> 5;
  const int wid  = blockIdx.x * 8 + wave;   // (N/32)*(KCH/64) = 1024*16 = 16384 waves
  const int n0 = (wid >> 4) * 32;
  const int k0 = (wid & 15) * 64;
  v8f acc[2][4] = {};
  for (int kk = 0; kk < FC0; kk += 32) {
    if (kk + 32 < FC0) {  // software prefetch of next A k-slice (global_prefetch_b8)
      __builtin_prefetch(hb + (size_t)(n0 + (lane & 15)) * FC0 + kk + 32, 0, 1);
      __builtin_prefetch(hb + (size_t)(n0 + 16 + (lane & 15)) * FC0 + kk + 32, 0, 1);
    }
    v16bf a0 = load_afrag(hb, FC0, n0,      kk, lane);
    v16bf a1 = load_afrag(hb, FC0, n0 + 16, kk, lane);
#pragma unroll
    for (int t = 0; t < 4; ++t) {
      v16bf b = load_bfrag_w(w2b, FC0, kk, k0 + 16 * t, lane);
      acc[0][t] = __builtin_amdgcn_wmma_f32_16x16x32_bf16(false, a0, false, b, (short)0, acc[0][t], false, false);
      acc[1][t] = __builtin_amdgcn_wmma_f32_16x16x32_bf16(false, a1, false, b, (short)0, acc[1][t], false, false);
    }
  }
  const int half = lane >> 4, ncol = lane & 15;
#pragma unroll
  for (int t = 0; t < 4; ++t) {
    int kch = k0 + 16 * t + ncol;
    float s = scale2[kch], bi = bias2[kch];
#pragma unroll
    for (int r = 0; r < 2; ++r) {
#pragma unroll
      for (int v = 0; v < 8; ++v) {
        int row = n0 + 16 * r + v + 8 * half;
        att[(size_t)row * KCH + kch] = fmaxf(acc[r][t][v] * s + bi, 0.0f);
      }
    }
  }
}

// ------------- segment max (att >= 0 so int-bit atomicMax with init 0 is exact) ------
__global__ void k_segmax(const float* __restrict__ att, const int* __restrict__ off,
                         float* __restrict__ mx) {
  const int k  = blockIdx.x * 256 + threadIdx.x;  // gridDim.x = KCH/256
  const int n0 = blockIdx.y * 256;                // gridDim.y = N/256
  int s = 0;
  while (n0 >= off[s + 1]) ++s;
  float m = 0.0f;
  bool any = false;
  for (int r = 0; r < 256; ++r) {
    int n = n0 + r;
    if (n >= off[s + 1]) {
      if (any) atomicMax((int*)&mx[s * KCH + k], __float_as_int(m));
      do { ++s; } while (n >= off[s + 1]);
      m = 0.0f; any = false;
    }
    m = fmaxf(m, att[(size_t)n * KCH + k]);
    any = true;
  }
  if (any) atomicMax((int*)&mx[s * KCH + k], __float_as_int(m));
}

// ------------- segment sum of exp(att - mx) ------------------------------------------
__global__ void k_segsum(const float* __restrict__ att, const int* __restrict__ off,
                         const float* __restrict__ mx, float* __restrict__ denom) {
  const int k  = blockIdx.x * 256 + threadIdx.x;
  const int n0 = blockIdx.y * 256;
  int s = 0;
  while (n0 >= off[s + 1]) ++s;
  float mv = mx[s * KCH + k];
  float acc = 0.0f;
  bool any = false;
  for (int r = 0; r < 256; ++r) {
    int n = n0 + r;
    if (n >= off[s + 1]) {
      if (any) atomicAdd(&denom[s * KCH + k], acc);
      do { ++s; } while (n >= off[s + 1]);
      mv = mx[s * KCH + k];
      acc = 0.0f; any = false;
    }
    acc += __expf(att[(size_t)n * KCH + k] - mv);
    any = true;
  }
  if (any) atomicAdd(&denom[s * KCH + k], acc);
}

// ------------- pooling: pooled[b,k,c] = (1/len) * sum_n w[n,k] x[n,c]  via WMMA -------
// Block = (segment b, 16-channel k-tile); 8 waves split the point chunks; LDS reduce.
__global__ void k_pool(const float* __restrict__ att, const unsigned short* __restrict__ xb,
                       const int* __restrict__ off, const float* __restrict__ mx,
                       const float* __restrict__ denom, const float* __restrict__ inv_len,
                       float* __restrict__ pooled) {
  __shared__ float red[8][16][32];
  const int lane = threadIdx.x & 31;
  const int wave = threadIdx.x >> 5;
  const int b  = blockIdx.x >> 6;   // gridDim.x = BSEG * (KCH/16)
  const int kt = blockIdx.x & 63;
  const int k0 = kt * 16;
  const int p0 = off[b], p1 = off[b + 1];
  const int half = lane >> 4, lidx = lane & 15;
  const int kch = k0 + lidx;                     // fixed channel per lane for A frag
  const float mv   = mx[b * KCH + kch];
  const float invd = 1.0f / denom[b * KCH + kch];
  v8f acc0 = {}, acc1 = {};
  const int nchunks = (p1 - p0 + 31) >> 5;
  for (int c = wave; c < nchunks; c += 8) {
    const int pb = p0 + c * 32;
    FragBF fa;                                   // A[m=kch, kk=point] = softmax weight
#pragma unroll
    for (int j = 0; j < 16; ++j) {
      int kidx = (j < 8) ? (8 * half + j) : (16 + 8 * half + (j - 8));
      int p = pb + kidx;
      float wv = 0.0f;
      if (p < p1) wv = __expf(att[(size_t)p * KCH + kch] - mv) * invd;
      fa.u[j] = f2bf_bits(wv);
    }
    FragBF fb0, fb1;                             // B[kk=point, n=c] = x
#pragma unroll
    for (int j = 0; j < 16; ++j) {
      int p = pb + ((j < 8) ? (8 * half + j) : (16 + 8 * half + (j - 8)));
      unsigned short x0 = 0, x1 = 0;
      if (p < p1) {
        x0 = xb[(size_t)p * C_IN + lidx];
        x1 = xb[(size_t)p * C_IN + 16 + lidx];
      }
      fb0.u[j] = x0; fb1.u[j] = x1;
    }
    acc0 = __builtin_amdgcn_wmma_f32_16x16x32_bf16(false, fa.v, false, fb0.v, (short)0, acc0, false, false);
    acc1 = __builtin_amdgcn_wmma_f32_16x16x32_bf16(false, fa.v, false, fb1.v, (short)0, acc1, false, false);
  }
#pragma unroll
  for (int v = 0; v < 8; ++v) {                  // D: lane=col c, VGPR v -> row m
    red[wave][v + 8 * half][lidx]      = acc0[v];
    red[wave][v + 8 * half][16 + lidx] = acc1[v];
  }
  __syncthreads();
  const float il = inv_len[b];
  for (int i = threadIdx.x; i < 512; i += 256) {
    int m = i >> 5, cc = i & 31;
    float sum = 0.0f;
#pragma unroll
    for (int w = 0; w < 8; ++w) sum += red[w][m][cc];
    pooled[((size_t)b * KCH + (k0 + m)) * C_IN + cc] = sum * il;
  }
}

// ------------- final FC: racc += out2 @ fcw^T  via f32 WMMA (fcw read once, native f32)
__global__ void k_fc(const float* __restrict__ pooled, const float* __restrict__ fcw,
                     float* __restrict__ racc) {
  const int lane = threadIdx.x & 31;
  const int wave = threadIdx.x >> 5;
  const int wid  = blockIdx.x * 8 + wave;   // 16 out-tiles * 64 splits = 1024 waves
  const int o0 = (wid >> 6) * 16;
  const int sp = wid & 63;
  const int kk0 = sp * 512, kk1 = kk0 + 512;
  const int half = lane >> 4, lidx = lane & 15;
  v8f acc = {};
  for (int kk = kk0; kk < kk1; kk += 4) {
    // stream-prefetch the fcw row one cacheline (32 floats) ahead
    if (((kk >> 2) & 7) == 0 && kk + 32 < kk1)
      __builtin_prefetch(fcw + (size_t)(o0 + lidx) * (KCH * C_IN) + kk + 32, 0, 1);
    v2f a, b;
#pragma unroll
    for (int j = 0; j < 2; ++j) {
      int kx = kk + 2 * half + j;           // A 16x4 f32: elem j -> k = 2*half + j
      a[j] = pooled[(size_t)lidx * (KCH * C_IN) + kx];
      b[j] = fcw[(size_t)(o0 + lidx) * (KCH * C_IN) + kx];
    }
    acc = __builtin_amdgcn_wmma_f32_16x16x4_f32(false, a, false, b, (short)0, acc, false, false);
  }
#pragma unroll
  for (int v = 0; v < 8; ++v) {
    int m = v + 8 * half;                   // batch row
    atomicAdd(&racc[m * FC0 + o0 + lidx], acc[v]);
  }
}

// ------------- affine3 + row-wise L2 normalize ---------------------------------------
__global__ void k_final(const float* __restrict__ racc, const float* __restrict__ scale3,
                        const float* __restrict__ bias3, float* __restrict__ out) {
  __shared__ float ssum[256];
  const int b = blockIdx.x, f = threadIdx.x;
  float v = racc[b * FC0 + f] * scale3[f] + bias3[f];
  ssum[f] = v * v;
  __syncthreads();
  for (int s = 128; s > 0; s >>= 1) {
    if (f < s) ssum[f] += ssum[f + s];
    __syncthreads();
  }
  float nrm = fmaxf(sqrtf(ssum[0]), 1e-12f);
  out[b * FC0 + f] = v / nrm;
}

extern "C" void kernel_launch(void* const* d_in, const int* in_sizes, int n_in,
                              void* d_out, int out_size, void* d_ws, size_t ws_size,
                              hipStream_t stream) {
  const float* x      = (const float*)d_in[0];
  const int*   length = (const int*)  d_in[1];
  const float* w1  = (const float*)d_in[2];
  const float* b1  = (const float*)d_in[3];
  const float* g1  = (const float*)d_in[4];
  const float* be1 = (const float*)d_in[5];
  const float* m1  = (const float*)d_in[6];
  const float* v1  = (const float*)d_in[7];
  const float* w2  = (const float*)d_in[8];
  const float* b2  = (const float*)d_in[9];
  const float* g2  = (const float*)d_in[10];
  const float* be2 = (const float*)d_in[11];
  const float* m2  = (const float*)d_in[12];
  const float* v2  = (const float*)d_in[13];
  const float* fcw = (const float*)d_in[14];
  const float* fcb = (const float*)d_in[15];
  const float* g3  = (const float*)d_in[16];
  const float* be3 = (const float*)d_in[17];
  const float* m3  = (const float*)d_in[18];
  const float* v3  = (const float*)d_in[19];
  float* out = (float*)d_out;

  char* ws = (char*)d_ws;
  size_t o = 0;
  auto alloc = [&](size_t bytes) -> void* {
    void* p = ws + o;
    o = (o + bytes + 255) & ~(size_t)255;
    return p;
  };
  int*   off     = (int*)  alloc(17 * 4);
  float* inv_len = (float*)alloc(16 * 4);
  float* scale1  = (float*)alloc(FC0 * 4);
  float* bias1   = (float*)alloc(FC0 * 4);
  float* scale2  = (float*)alloc(KCH * 4);
  float* bias2   = (float*)alloc(KCH * 4);
  float* scale3  = (float*)alloc(FC0 * 4);
  float* bias3   = (float*)alloc(FC0 * 4);
  // mx, denom, racc are contiguous (sizes are multiples of 256B) -> one zeroing pass
  float* mx      = (float*)alloc((size_t)BSEG * KCH * 4);
  float* denom   = (float*)alloc((size_t)BSEG * KCH * 4);
  float* racc    = (float*)alloc((size_t)BSEG * FC0 * 4);
  unsigned short* xb  = (unsigned short*)alloc((size_t)N_PTS * C_IN * 2);
  unsigned short* w1b = (unsigned short*)alloc((size_t)FC0 * C_IN * 2);
  unsigned short* w2b = (unsigned short*)alloc((size_t)KCH * FC0 * 2);
  unsigned short* hb  = (unsigned short*)alloc((size_t)N_PTS * FC0 * 2);
  float* pooled  = (float*)alloc((size_t)BSEG * KCH * C_IN * 4);
  float* att     = (float*)alloc((size_t)N_PTS * KCH * 4);

  k_prep<<<1, 1024, 0, stream>>>(length, b1, g1, be1, m1, v1, b2, g2, be2, m2, v2,
                                 fcb, g3, be3, m3, v3,
                                 scale1, bias1, scale2, bias2, scale3, bias3, off, inv_len);
  k_cvt<<<(N_PTS * C_IN + 255) / 256, 256, 0, stream>>>(x,  xb,  N_PTS * C_IN);
  k_cvt<<<(FC0 * C_IN + 255) / 256,   256, 0, stream>>>(w1, w1b, FC0 * C_IN);
  k_cvt<<<(KCH * FC0 + 255) / 256,    256, 0, stream>>>(w2, w2b, KCH * FC0);
  const int nz = BSEG * KCH * 2 + BSEG * FC0;   // mx + denom + racc
  k_zero<<<(nz + 255) / 256, 256, 0, stream>>>(mx, nz);

  k_gemm1<<<(N_PTS / 16) * (FC0 / 16) / 8, 256, 0, stream>>>(xb, w1b, scale1, bias1, hb);
  k_gemm2<<<(N_PTS / 32) * (KCH / 64) / 8, 256, 0, stream>>>(hb, w2b, scale2, bias2, att);

  dim3 gs(KCH / 256, N_PTS / 256);
  k_segmax<<<gs, 256, 0, stream>>>(att, off, mx);
  k_segsum<<<gs, 256, 0, stream>>>(att, off, mx, denom);

  k_pool<<<BSEG * (KCH / 16), 256, 0, stream>>>(att, xb, off, mx, denom, inv_len, pooled);
  k_fc<<<(16 * 64) / 8, 256, 0, stream>>>(pooled, fcw, racc);
  k_final<<<BSEG, 256, 0, stream>>>(racc, scale3, bias3, out);
}
